// PairDrift_20461224198619
// MI455X (gfx1250) — matrix operand order repaired and checked by reading, not compiled, not run on hardware
//
#include <hip/hip_runtime.h>
#include <hip/hip_bf16.h>

#define BATCH 4
#define NPTS  512
#define DIM   3
#define HID   128

typedef _Float16 h2   __attribute__((ext_vector_type(2)));
typedef _Float16 h8   __attribute__((ext_vector_type(8)));
typedef _Float16 v16h __attribute__((ext_vector_type(16)));
typedef float    v8f  __attribute__((ext_vector_type(8)));

__device__ __forceinline__ float ht(float v) {
    return fminf(1.0f, fmaxf(-1.0f, v));
}

// One block per (b, j). Fuses: pair construction, p = ht(pairs@Wp1+bp1),
// psum1 = sum_i p, p2 = ht(p@Wpl+bpl) via f16 WMMA (f32 accum), psum2 = sum_i p2,
// then the whole single-stream epilogue (combine1 -> ht -> combine2 + skip).
// Double-buffered p-tile + LDS-resident x rows => one barrier per i-chunk.
__global__ __launch_bounds__(256) void PairDrift_fused_kernel(
    const float* __restrict__ x,
    const float* __restrict__ Ws1,  const float* __restrict__ bs1,
    const float* __restrict__ Wp1,  const float* __restrict__ bp1,
    const float* __restrict__ Wc1s, const float* __restrict__ bc1s,
    const float* __restrict__ Wc1p, const float* __restrict__ bc1p,
    const float* __restrict__ Wpl,  const float* __restrict__ bpl,
    const float* __restrict__ Wc2s, const float* __restrict__ bc2s,
    const float* __restrict__ Wc2p, const float* __restrict__ bc2p,
    const float* __restrict__ Wskip,
    float* __restrict__ out)
{
    __shared__ __align__(16) _Float16 ldsP[2][32 * 136]; // double-buffered 32x128(+pad) f16
    __shared__ float ldsX[NPTS * DIM];                   // all x rows of this batch (6 KB)
    __shared__ float red[512];                           // generic reduction scratch
    __shared__ float s1L[HID], s2L[HID], ps1L[HID], ps2L[HID];

    const int t     = threadIdx.x;
    const int lane  = t & 31;
    const int wave  = t >> 5;          // 0..7 : owns output cols [16w,16w+16)
    const int b     = blockIdx.y;
    const int j     = blockIdx.x;

    // p-generation mapping: thread owns an even/odd column pair (h0, h0+1)
    // and rows {rbase, rbase+4, ..., rbase+28} of the 32-row chunk.
    const int h0    = (t & 63) * 2;
    const int rbase = t >> 6;          // 0..3

    // stage all x rows for this b into LDS once (read-only afterwards)
    for (int u = t; u < NPTS * DIM; u += 256)
        ldsX[u] = x[b * NPTS * DIM + u];

    // x row j (tiny, L2 resident)
    const float xj0 = x[(b * NPTS + j) * DIM + 0];
    const float xj1 = x[(b * NPTS + j) * DIM + 1];
    const float xj2 = x[(b * NPTS + j) * DIM + 2];

    // Wp1 columns h0 / h0+1 (negated so a_j - x.w lowers as a pure FMA chain)
    const float w0a = Wp1[0 * HID + h0],     w0b = Wp1[0 * HID + h0 + 1];
    const float w1a = Wp1[1 * HID + h0],     w1b = Wp1[1 * HID + h0 + 1];
    const float w2a = Wp1[2 * HID + h0],     w2b = Wp1[2 * HID + h0 + 1];
    const float aja = xj0 * w0a + xj1 * w1a + xj2 * w2a + bp1[h0];
    const float ajb = xj0 * w0b + xj1 * w1b + xj2 * w2b + bp1[h0 + 1];
    const float nw0a = -w0a, nw1a = -w1a, nw2a = -w2a;
    const float nw0b = -w0b, nw1b = -w1b, nw2b = -w2b;

    // ---- B fragments: Wpl[:, 16w + n] as f16, resident in VGPRs all kernel ----
    const int n_loc  = lane & 15;
    const int khalf  = lane >> 4;
    const int n_glob = wave * 16 + n_loc;
    v16h Bfrag[4];
    #pragma unroll
    for (int kf = 0; kf < 4; ++kf) {
        #pragma unroll
        for (int u = 0; u < 16; ++u) {
            const int k = kf * 32 + (u < 8 ? khalf * 8 + u : 16 + khalf * 8 + (u - 8));   // 16-bit B layout: lane=col, K in table order (8-blocks by half-wave)
            Bfrag[kf][u] = (_Float16)Wpl[k * HID + n_glob];
        }
    }
    const float bpl_lane = bpl[n_glob];

    float psum1a = 0.0f, psum1b = 0.0f;   // partials of sum_i p[i, h0] / p[i, h0+1]
    float psum2a = 0.0f, psum2b = 0.0f;   // partials of sum_i p2[i, n_glob] (two row tiles)

    const int arow = lane & 15;           // A layout: lane = row within tile

    __syncthreads();                      // ldsX visible to everyone

    // ================= main i loop: 16 chunks of 32 rows, 1 barrier each =============
    int par = 0;
    for (int i0 = 0; i0 < NPTS; i0 += 32) {
        _Float16* __restrict__ tile = &ldsP[par][0];

        // build f16 p-tile (packed dword stores) + f32 psum1 accumulation
        #pragma unroll
        for (int rr = 0; rr < 8; ++rr) {
            const int il = rr * 4 + rbase;
            const float xa = ldsX[(i0 + il) * 3 + 0];
            const float xb = ldsX[(i0 + il) * 3 + 1];
            const float xc = ldsX[(i0 + il) * 3 + 2];
            const float pa = ht(fmaf(nw0a, xa, fmaf(nw1a, xb, fmaf(nw2a, xc, aja))));
            const float pb = ht(fmaf(nw0b, xa, fmaf(nw1b, xb, fmaf(nw2b, xc, ajb))));
            psum1a += pa;
            psum1b += pb;
            h2 pk; pk[0] = (_Float16)pa; pk[1] = (_Float16)pb;
            *(h2*)&tile[il * 136 + h0] = pk;           // one ds_store_b32
        }
        __syncthreads();                               // tile ready; prev buffer now free

        // ---- preload all 8 A fragments, then two independent 4-WMMA chains ----
        v16h A0[4], A1[4];
        #pragma unroll
        for (int kf = 0; kf < 4; ++kf) {
            const _Float16* rp0 = &tile[arow * 136 + kf * 32 + khalf * 8];
            const _Float16* rp1 = rp0 + 16 * 136;
            h8 lo0 = *(const h8*)rp0;                  // K = kf*32 + khalf*8 + 0..7
            h8 hi0 = *(const h8*)(rp0 + 16);           // K = kf*32 + 16 + khalf*8 + 0..7
            h8 lo1 = *(const h8*)rp1;
            h8 hi1 = *(const h8*)(rp1 + 16);
            A0[kf] = __builtin_shufflevector(lo0, hi0, 0,1,2,3,4,5,6,7,8,9,10,11,12,13,14,15);
            A1[kf] = __builtin_shufflevector(lo1, hi1, 0,1,2,3,4,5,6,7,8,9,10,11,12,13,14,15);
        }
        v8f acc0 = {};
        v8f acc1 = {};
        #pragma unroll
        for (int kf = 0; kf < 4; ++kf) {
            acc0 = __builtin_amdgcn_wmma_f32_16x16x32_f16(
                       false, A0[kf], false, Bfrag[kf], (short)0, acc0, false, false);
            acc1 = __builtin_amdgcn_wmma_f32_16x16x32_f16(
                       false, A1[kf], false, Bfrag[kf], (short)0, acc1, false, false);
        }
        // p2 = ht(acc + bpl); fold the sum over this chunk's 32 rows immediately
        #pragma unroll
        for (int r = 0; r < 8; ++r) {
            psum2a += ht(acc0[r] + bpl_lane);
            psum2b += ht(acc1[r] + bpl_lane);
        }
        par ^= 1;
    }

    // ================= cross-thread reductions of the i-sums =================
    __syncthreads();
    red[t]       = psum1a;
    red[256 + t] = psum1b;
    __syncthreads();
    if (t < 128) {                                     // 4 threads per h column
        const int base = (t & 1) * 256 + (t >> 1);
        ps1L[t] = (red[base] + red[base + 64]) + (red[base + 128] + red[base + 192]);
    }
    __syncthreads();
    red[t] = psum2a + psum2b;                          // same column for both tiles
    __syncthreads();
    if (t < 128) {                                     // h = 16*(t>>4) + (t&15)
        const int base = (t >> 4) * 32 + (t & 15);
        ps2L[t] = red[base] + red[base + 16];          // lane L and L+16 halves
    }
    // s1 = ht(x @ Ws1 + bs1)
    if (t < 128)
        s1L[t] = ht(xj0 * Ws1[t] + xj1 * Ws1[HID + t] + xj2 * Ws1[2 * HID + t] + bs1[t]);
    __syncthreads();

    // combine1: s2 = ht(s1@Wc1s + bc1s + psum1@Wc1p + N*bc1p)  (split-K over thread halves)
    {
        const int col = t & 127;
        const int kb  = (t >> 7) * 64;
        float part = 0.0f;
        for (int k = kb; k < kb + 64; ++k)
            part += s1L[k] * Wc1s[k * HID + col] + ps1L[k] * Wc1p[k * HID + col];
        red[t] = part;
    }
    __syncthreads();
    if (t < 128)
        s2L[t] = ht(red[t] + red[t + 128] + bc1s[t] + (float)NPTS * bc1p[t]);
    __syncthreads();

    // combine2 + skip: out = s2@Wc2s + bc2s + psum2@Wc2p + N*bc2p + x@Wskip^T
    if (t < 128) {
        #pragma unroll
        for (int d = 0; d < DIM; ++d)
            red[t * DIM + d] = s2L[t] * Wc2s[t * DIM + d] + ps2L[t] * Wc2p[t * DIM + d];
    }
    __syncthreads();
    if (t < DIM) {
        float acc = bc2s[t] + (float)NPTS * bc2p[t];
        for (int k = 0; k < HID; ++k) acc += red[k * DIM + t];   // fixed order: deterministic
        acc += xj0 * Wskip[t * DIM + 0] + xj1 * Wskip[t * DIM + 1] + xj2 * Wskip[t * DIM + 2];
        out[(b * NPTS + j) * DIM + t] = acc;
    }
}

extern "C" void kernel_launch(void* const* d_in, const int* in_sizes, int n_in,
                              void* d_out, int out_size, void* d_ws, size_t ws_size,
                              hipStream_t stream) {
    const float* x     = (const float*)d_in[0];
    const float* Ws1   = (const float*)d_in[1];
    const float* bs1   = (const float*)d_in[2];
    const float* Wp1   = (const float*)d_in[3];
    const float* bp1   = (const float*)d_in[4];
    const float* Wc1s  = (const float*)d_in[5];
    const float* bc1s  = (const float*)d_in[6];
    const float* Wc1p  = (const float*)d_in[7];
    const float* bc1p  = (const float*)d_in[8];
    const float* Wpl   = (const float*)d_in[9];
    const float* bpl   = (const float*)d_in[10];
    const float* Wc2s  = (const float*)d_in[11];
    const float* bc2s  = (const float*)d_in[12];
    const float* Wc2p  = (const float*)d_in[13];
    const float* bc2p  = (const float*)d_in[14];
    const float* Wskip = (const float*)d_in[15];

    dim3 grid(NPTS, BATCH);   // 2048 blocks, one (b, j) each
    PairDrift_fused_kernel<<<grid, 256, 0, stream>>>(
        x, Ws1, bs1, Wp1, bp1, Wc1s, bc1s, Wc1p, bc1p,
        Wpl, bpl, Wc2s, bc2s, Wc2p, bc2p, Wskip, (float*)d_out);
}
